// MyModel_52639119180344
// MI455X (gfx1250) — matrix operand order repaired and checked by reading, not compiled, not run on hardware
//
#include <hip/hip_runtime.h>
#include <cstdint>

typedef __attribute__((ext_vector_type(2)))  float    v2f;
typedef __attribute__((ext_vector_type(8)))  float    v8f;
typedef __attribute__((ext_vector_type(16))) __bf16   v16bf;
typedef __attribute__((ext_vector_type(8)))  __bf16   v8bf;
typedef __attribute__((ext_vector_type(4)))  uint32_t v4u;

#define HID    512
#define EMB    256
#define VOCAB  32000
#define BATCH  32
#define SEQ    256
#define G4     2048          /* 4*HID */
#define NROWS  8192          /* B*S   */

// ---------- helpers ----------
__device__ __forceinline__ uint16_t f2bf_bits(float f) {
    uint32_t u = __float_as_uint(f);
    return (uint16_t)((u + 0x7FFFu + ((u >> 16) & 1u)) >> 16);   // RNE
}
__device__ __forceinline__ float sigmoidf_(float x) {
    x = fminf(fmaxf(x, -20.f), 20.f);
    return 1.0f / (1.0f + __expf(-x));
}
__device__ __forceinline__ float tanhf_(float x) {
    x = fminf(fmaxf(x, -15.f), 15.f);
    float e = __expf(2.0f * x);
    return (e - 1.0f) / (e + 1.0f);
}

// ---------- one-time layout transforms ----------
// W[N][K] f32 -> fragment-paired layout P (as v2f):
//   P[(kq*2 + half)*N + n] = { W[n][4kq+half], W[n][4kq+half+2] }
// so an f32-WMMA B fragment (K-step kk, lane half) is ONE b64 load.
__global__ void packB_f32(const float* __restrict__ W, v2f* __restrict__ P, int N, int K) {
    int idx = blockIdx.x * blockDim.x + threadIdx.x;      // over N*K/2
    if (idx >= N * (K / 2)) return;
    int kq   = idx / (2 * N);
    int rem  = idx % (2 * N);
    int half = rem / N;
    int n    = rem % N;
    v2f v;
    v.x = W[(size_t)n * K + 4 * kq + half];
    v.y = W[(size_t)n * K + 4 * kq + half + 2];
    P[idx] = v;
}

// W_out[32000][512] f32 -> bf16 WMMA B fragments, per-lane contiguous:
//   Bp[(((nt*16 + kb)*32 + l)*8 + j] = pack(bf16 W[nt*16+(l&15)][kb*32+(l>>4)*16+2j],
//                                           bf16 W[.. ][.. +1])
__global__ void pack_wout_frag(const float* __restrict__ W, uint32_t* __restrict__ Bp) {
    int idx = blockIdx.x * blockDim.x + threadIdx.x;      // over 2000*16*32*8 = 8,192,000
    if (idx >= (VOCAB / 16) * 16 * 32 * 8) return;
    int j  = idx & 7;
    int l  = (idx >> 3) & 31;
    int kb = (idx >> 8) & 15;
    int nt = idx >> 12;
    int n  = nt * 16 + (l & 15);
    int K  = kb * 32 + (l >> 4) * 16 + 2 * j;
    uint32_t lo = f2bf_bits(W[(size_t)n * HID + K]);
    uint32_t hi = f2bf_bits(W[(size_t)n * HID + K + 1]);
    Bp[idx] = lo | (hi << 16);
}

__global__ void zero4(float* a, float* b, float* c, float* d) {
    int i = blockIdx.x * blockDim.x + threadIdx.x;
    if (i < BATCH * HID) { a[i] = 0.f; b[i] = 0.f; c[i] = 0.f; d[i] = 0.f; }
}

// ---------- fp32 WMMA pre-activation GEMMs ----------
// pre[r][n] = A(r)·W[:,n] + bi[n] + bh[n], rows r = t*32+b, N = 2048.
// grid = (512 Mtiles, 8), block = 128 (4 waves), each wave does a 16x64 tile.

__global__ void pre_gemm_embed(const int* __restrict__ tok, const float* __restrict__ emb,
                               const v2f* __restrict__ WP,    // packed [EMB/4][2][G4]
                               const float* __restrict__ bi, const float* __restrict__ bh,
                               float* __restrict__ pre) {
    const int lane = threadIdx.x & 31;
    const int wave = threadIdx.x >> 5;
    const int mb   = blockIdx.x * 16;
    const int ng   = (blockIdx.y * 4 + wave) * 64;
    const int m    = mb + (lane & 15);
    const int t    = m >> 5, b = m & 31;
    const float* __restrict__ arow = emb + (size_t)tok[b * SEQ + t] * EMB;
    const int ksel = (lane >> 4) * 2;          // A: lanes 0-15 K={0,1}, 16-31 K={2,3}
    const int kbo  = (lane >> 4);              // B: half select
    v8f acc[4] = {};
#pragma unroll 1
    for (int kk = 0; kk < EMB; kk += 4) {
        v2f a = *(const v2f*)(arow + kk + ksel);
        const v2f* __restrict__ brow = WP + (size_t)((kk >> 2) * 2 + kbo) * G4 + ng + (lane & 15);
#pragma unroll
        for (int tt = 0; tt < 4; tt++) {
            v2f bfr = brow[tt * 16];
            acc[tt] = __builtin_amdgcn_wmma_f32_16x16x4_f32(false, a, false, bfr,
                                                            (short)0, acc[tt], false, false);
        }
    }
#pragma unroll
    for (int tt = 0; tt < 4; tt++)
        for (int r = 0; r < 8; r++) {
            int row = mb + (lane >> 4) * 8 + r;
            int col = ng + tt * 16 + (lane & 15);
            pre[(size_t)row * G4 + col] = acc[tt][r] + bi[col] + bh[col];
        }
}

__global__ void pre_gemm(const float* __restrict__ A,          // [NROWS][HID]
                         const v2f* __restrict__ WP,           // packed [HID/4][2][G4]
                         const float* __restrict__ bi, const float* __restrict__ bh,
                         float* __restrict__ pre) {
    const int lane = threadIdx.x & 31;
    const int wave = threadIdx.x >> 5;
    const int mb   = blockIdx.x * 16;
    const int ng   = (blockIdx.y * 4 + wave) * 64;
    const float* __restrict__ arow = A + (size_t)(mb + (lane & 15)) * HID;
    const int ksel = (lane >> 4) * 2;
    const int kbo  = (lane >> 4);
    v8f acc[4] = {};
#pragma unroll 1
    for (int kk = 0; kk < HID; kk += 4) {
        v2f a = *(const v2f*)(arow + kk + ksel);
        const v2f* __restrict__ brow = WP + (size_t)((kk >> 2) * 2 + kbo) * G4 + ng + (lane & 15);
#pragma unroll
        for (int tt = 0; tt < 4; tt++) {
            v2f bfr = brow[tt * 16];
            acc[tt] = __builtin_amdgcn_wmma_f32_16x16x4_f32(false, a, false, bfr,
                                                            (short)0, acc[tt], false, false);
        }
    }
#pragma unroll
    for (int tt = 0; tt < 4; tt++)
        for (int r = 0; r < 8; r++) {
            int row = mb + (lane >> 4) * 8 + r;
            int col = ng + tt * 16 + (lane & 15);
            pre[(size_t)row * G4 + col] = acc[tt][r] + bi[col] + bh[col];
        }
}

// ---------- one LSTM timestep (fp32 WMMA + fused gates) ----------
// grid = 32 (16-col groups), block = 64 (2 waves: row tiles 0-15 / 16-31).
// z = pre_t + h_in · W_hh^T ; i,f,g,o tiles live at col, col+512, +1024, +1536.
__global__ void lstm_step(const float* __restrict__ pre_t,   // [32][2048]
                          const v2f* __restrict__ WhhP,      // packed [HID/4][2][G4]
                          const float* __restrict__ h_in,    // [32][512]
                          float* __restrict__ h_out,         // [32][512]
                          float* __restrict__ c,             // [32][512]
                          float* __restrict__ y_t) {         // [32][512]
    const int lane = threadIdx.x & 31;
    const int wave = threadIdx.x >> 5;
    const int mb   = wave * 16;
    const int colI = blockIdx.x * 16 + (lane & 15);
    v8f acc[4];
#pragma unroll
    for (int g = 0; g < 4; g++)
        for (int r = 0; r < 8; r++) {
            int row = mb + (lane >> 4) * 8 + r;
            acc[g][r] = pre_t[(size_t)row * G4 + g * HID + colI];
        }
    const float* __restrict__ arow = h_in + (size_t)(mb + (lane & 15)) * HID;
    const int ksel = (lane >> 4) * 2;
    const int kbo  = (lane >> 4);
#pragma unroll 1
    for (int kk = 0; kk < HID; kk += 4) {
        v2f a = *(const v2f*)(arow + kk + ksel);
        const v2f* __restrict__ brow = WhhP + (size_t)((kk >> 2) * 2 + kbo) * G4 + colI;
#pragma unroll
        for (int g = 0; g < 4; g++) {
            v2f bfr = brow[g * HID];
            acc[g] = __builtin_amdgcn_wmma_f32_16x16x4_f32(false, a, false, bfr,
                                                           (short)0, acc[g], false, false);
        }
    }
#pragma unroll
    for (int r = 0; r < 8; r++) {
        int row = mb + (lane >> 4) * 8 + r;
        float ig = sigmoidf_(acc[0][r]);
        float fg = sigmoidf_(acc[1][r]);
        float gg = tanhf_(acc[2][r]);
        float og = sigmoidf_(acc[3][r]);
        size_t idx = (size_t)row * HID + colI;
        float cn = fg * c[idx] + ig * gg;
        float hn = og * tanhf_(cn);
        c[idx] = cn; h_out[idx] = hn; y_t[idx] = hn;
    }
}

// ---------- y1 -> bf16 with [S,B]->[B,S] row remap ----------
__global__ void convert_y1(const float* __restrict__ y1, uint16_t* __restrict__ ybf) {
    int idx = blockIdx.x * blockDim.x + threadIdx.x;      // over 8192*512
    if (idx >= NROWS * HID) return;
    int r = idx / HID, k = idx % HID;                     // r = b*256 + t (logits order)
    int b = r >> 8, t = r & 255;
    ybf[idx] = f2bf_bits(y1[(size_t)(t * BATCH + b) * HID + k]);
}

// ---------- big vocab GEMM: bf16 WMMA 16x16x32 ----------
// grid = (125, 256), block = 128 (4 waves), wave = 32(M)x64(N) tile (128 WMMA/wave).
// B fragments are pre-swizzled per-lane-contiguous: two b128 loads per fragment.
__global__ void out_gemm_bf16(const __bf16* __restrict__ Abf,     // [8192][512]
                              const uint32_t* __restrict__ Bp,    // fragment layout
                              const float* __restrict__ bias,     // [32000]
                              float* __restrict__ out) {          // [8192][32000]
    const int lane = threadIdx.x & 31;
    const int wave = threadIdx.x >> 5;
    const int mb   = blockIdx.y * 32;
    const int nb   = (blockIdx.x * 4 + wave) * 64;
    const int ntg  = nb >> 4;                             // first of 4 n-tiles
    const __bf16* __restrict__ arow0 = Abf + (size_t)(mb + (lane & 15)) * HID;
    const __bf16* __restrict__ arow1 = arow0 + 16 * HID;
    const int hoff = (lane >> 4) * 16;                    // lane halves: K 0-15 vs 16-31
    v8f acc[2][4] = {};
#pragma unroll 1
    for (int k0 = 0; k0 < HID; k0 += 32) {
        v16bf a0, a1;
        {
            v8bf lo = *(const v8bf*)(arow0 + k0 + hoff);
            v8bf hi = *(const v8bf*)(arow0 + k0 + 8 + hoff);
#pragma unroll
            for (int i = 0; i < 8; i++) { a0[i] = lo[i]; a0[8 + i] = hi[i]; }
        }
        {
            v8bf lo = *(const v8bf*)(arow1 + k0 + hoff);
            v8bf hi = *(const v8bf*)(arow1 + k0 + 8 + hoff);
#pragma unroll
            for (int i = 0; i < 8; i++) { a1[i] = lo[i]; a1[8 + i] = hi[i]; }
        }
        const int kb = k0 >> 5;
#pragma unroll
        for (int tt = 0; tt < 4; tt++) {
            const uint32_t* bp = Bp + ((((size_t)(ntg + tt) * 16 + kb) * 32 + lane) << 3);
            union { v16bf v; v4u u4[2]; } bfr;
            bfr.u4[0] = *(const v4u*)(bp);
            bfr.u4[1] = *(const v4u*)(bp + 4);
            acc[0][tt] = __builtin_amdgcn_wmma_f32_16x16x32_bf16(false, a0, false, bfr.v,
                                                                 (short)0, acc[0][tt], false, false);
            acc[1][tt] = __builtin_amdgcn_wmma_f32_16x16x32_bf16(false, a1, false, bfr.v,
                                                                 (short)0, acc[1][tt], false, false);
        }
    }
#pragma unroll
    for (int mt = 0; mt < 2; mt++)
        for (int tt = 0; tt < 4; tt++)
            for (int r = 0; r < 8; r++) {
                int row = mb + mt * 16 + (lane >> 4) * 8 + r;
                int col = nb + tt * 16 + (lane & 15);
                out[(size_t)row * VOCAB + col] = acc[mt][tt][r] + bias[col];
            }
}

// ---------- final h_n / c_n ----------
__global__ void copy_states(const float* h0, const float* h1,
                            const float* c0, const float* c1, float* out) {
    int i = blockIdx.x * blockDim.x + threadIdx.x;
    if (i >= BATCH * HID) return;
    size_t base = (size_t)NROWS * VOCAB;
    out[base + i]                    = h0[i];
    out[base + BATCH * HID + i]      = h1[i];
    out[base + 2 * BATCH * HID + i]  = c0[i];
    out[base + 3 * BATCH * HID + i]  = c1[i];
}

// =====================================================================
extern "C" void kernel_launch(void* const* d_in, const int* in_sizes, int n_in,
                              void* d_out, int out_size, void* d_ws, size_t ws_size,
                              hipStream_t stream) {
    const int*   tok  = (const int*)d_in[0];
    const float* emb  = (const float*)d_in[1];
    const float* Wih0 = (const float*)d_in[2];
    const float* Whh0 = (const float*)d_in[3];
    const float* bih0 = (const float*)d_in[4];
    const float* bhh0 = (const float*)d_in[5];
    const float* Wih1 = (const float*)d_in[6];
    const float* Whh1 = (const float*)d_in[7];
    const float* bih1 = (const float*)d_in[8];
    const float* bhh1 = (const float*)d_in[9];
    const float* Wout = (const float*)d_in[10];
    const float* bout = (const float*)d_in[11];
    float* out = (float*)d_out;

    char* p = (char*)d_ws;
    auto alloc = [&](size_t bytes) { void* r = (void*)p; p += (bytes + 255) & ~(size_t)255; return r; };
    v2f*      Wih0P = (v2f*)alloc((size_t)EMB * G4 * 4);
    v2f*      Whh0P = (v2f*)alloc((size_t)HID * G4 * 4);
    v2f*      Wih1P = (v2f*)alloc((size_t)HID * G4 * 4);
    v2f*      Whh1P = (v2f*)alloc((size_t)HID * G4 * 4);
    uint32_t* WoutP = (uint32_t*)alloc((size_t)(HID / 2) * VOCAB * 4);
    float*    pre   = (float*)alloc((size_t)NROWS * G4 * 4);     // reused for both layers
    float*    y0    = (float*)alloc((size_t)NROWS * HID * 4);
    float*    y1    = (float*)alloc((size_t)NROWS * HID * 4);
    uint16_t* y1bf  = (uint16_t*)alloc((size_t)NROWS * HID * 2);
    float*    hA0   = (float*)alloc(BATCH * HID * 4);
    float*    hB0   = (float*)alloc(BATCH * HID * 4);
    float*    cS0   = (float*)alloc(BATCH * HID * 4);
    float*    hA1   = (float*)alloc(BATCH * HID * 4);
    float*    hB1   = (float*)alloc(BATCH * HID * 4);
    float*    cS1   = (float*)alloc(BATCH * HID * 4);

    // one-time layout transforms
    packB_f32<<<(G4 * (EMB / 2) + 255) / 256, 256, 0, stream>>>(Wih0, Wih0P, G4, EMB);
    packB_f32<<<(G4 * (HID / 2) + 255) / 256, 256, 0, stream>>>(Whh0, Whh0P, G4, HID);
    packB_f32<<<(G4 * (HID / 2) + 255) / 256, 256, 0, stream>>>(Wih1, Wih1P, G4, HID);
    packB_f32<<<(G4 * (HID / 2) + 255) / 256, 256, 0, stream>>>(Whh1, Whh1P, G4, HID);
    pack_wout_frag<<<(VOCAB * (HID / 2) + 255) / 256, 256, 0, stream>>>(Wout, WoutP);
    zero4<<<(BATCH * HID + 255) / 256, 256, 0, stream>>>(hA0, cS0, hA1, cS1);

    // layer 0: pre-activations then 256 recurrent steps (h double-buffered)
    pre_gemm_embed<<<dim3(NROWS / 16, 8), 128, 0, stream>>>(tok, emb, Wih0P, bih0, bhh0, pre);
    for (int t = 0; t < SEQ; t++) {
        const float* hin = (t & 1) ? hB0 : hA0;
        float* hout      = (t & 1) ? hA0 : hB0;
        lstm_step<<<HID / 16, 64, 0, stream>>>(pre + (size_t)t * BATCH * G4, Whh0P,
                                               hin, hout, cS0,
                                               y0 + (size_t)t * BATCH * HID);
    }
    // layer 1
    pre_gemm<<<dim3(NROWS / 16, 8), 128, 0, stream>>>(y0, Wih1P, bih1, bhh1, pre);
    for (int t = 0; t < SEQ; t++) {
        const float* hin = (t & 1) ? hB1 : hA1;
        float* hout      = (t & 1) ? hA1 : hB1;
        lstm_step<<<HID / 16, 64, 0, stream>>>(pre + (size_t)t * BATCH * G4, Whh1P,
                                               hin, hout, cS1,
                                               y1 + (size_t)t * BATCH * HID);
    }
    // vocab projection (bf16 WMMA)
    convert_y1<<<(NROWS * HID + 255) / 256, 256, 0, stream>>>(y1, y1bf);
    out_gemm_bf16<<<dim3(VOCAB / 256, NROWS / 32), 128, 0, stream>>>(
        (const __bf16*)y1bf, WoutP, bout, out);
    // final states: SEQ is even, so the last write of each layer landed in the "A" buffers
    copy_states<<<(BATCH * HID + 255) / 256, 256, 0, stream>>>(hA0, hA1, cS0, cS1, out);
}
